// SwinTransformerBlock_9345848836114
// MI455X (gfx1250) — matrix operand order repaired and checked by reading, not compiled, not run on hardware
//
#include <hip/hip_runtime.h>

typedef __bf16 bf16;
typedef __attribute__((ext_vector_type(16))) __bf16 v16bf;
typedef __attribute__((ext_vector_type(8)))  float  v8f;
typedef int v4i __attribute__((vector_size(16)));

#define DEV __device__ __forceinline__

#ifndef __has_builtin
#define __has_builtin(x) 0
#endif

#if defined(__HIP_DEVICE_COMPILE__) && __has_builtin(__builtin_amdgcn_global_load_async_to_lds_b128)
#define HAVE_ASYNC_LDS 1
#else
#define HAVE_ASYNC_LDS 0
#if defined(__HIP_DEVICE_COMPILE__)
#warning "gfx1250 async-to-LDS builtin not available; falling back to direct LDS copies"
#endif
#endif

static constexpr int Hdim = 56, Wdim = 56, WS = 7, SHIFT = 3, NH = 12;
static constexpr int Bsz = 32, Cdim = 384, HID = 1536, Nwin = 49;
static constexpr int Lseq = Hdim * Wdim;                      // 3136
static constexpr int NWINS = Bsz * (Hdim / WS) * (Wdim / WS); // 2048
static constexpr int Mrows = NWINS * Nwin;                    // 100352 = 128*784
static constexpr int QKV = 3 * Cdim;                          // 1152
static constexpr int LDST = 40;                               // padded LDS row stride (elements)

union BfVec { v16bf v; uint4 q[2]; };

DEV v16bf zero_frag() {
  BfVec u; u.q[0] = make_uint4(0,0,0,0); u.q[1] = make_uint4(0,0,0,0); return u.v;
}
// 16 contiguous bf16 (32B): B-fragment pattern (lane n=l%16, K-half by l/16)
DEV v16bf load_bf16x16(const bf16* p) {
  BfVec u; u.q[0] = *(const uint4*)p; u.q[1] = *(const uint4*)(p + 8); return u.v;
}
// A-fragment: lane<16 -> K {0..15}; lane>=16 -> K {16..31}
DEV v16bf load_a_pair(const bf16* p) {
  BfVec u; u.q[0] = *(const uint4*)p; u.q[1] = *(const uint4*)(p + 8); return u.v;
}
DEV v8f wmma_bf16(v16bf a, v16bf b, v8f c) {
  return __builtin_amdgcn_wmma_f32_16x16x32_bf16(false, a, false, b, (short)0, c, false, false);
}
DEV void win_map(int rowg, int& t) { // window-token row -> spatial token (roll+partition map)
  int win = rowg / Nwin, n = rowg % Nwin;
  int batch = win >> 6, wrem = win & 63, wh = wrem >> 3, ww = wrem & 7;
  int i = n / WS, j = n % WS;
  int hh = (wh * WS + i + SHIFT) % Hdim;
  int wp = (ww * WS + j + SHIFT) % Wdim;
  t = batch * Lseq + hh * Wdim + wp;
}

// ---- async global->LDS copy (16B per call), with portable fallback ---------
DEV void copy16_async(const bf16* g, bf16* l) {
#if HAVE_ASYNC_LDS
  __builtin_amdgcn_global_load_async_to_lds_b128((v4i*)g, (v4i*)l, 0, 0);
#else
  *(uint4*)l = *(const uint4*)g;
#endif
}
template <int N> DEV void async_wait() {
#if HAVE_ASYNC_LDS
  asm volatile("s_wait_asynccnt %0" :: "i"(N) : "memory");
#endif
}
// copy one 128x32 bf16 k-slice (row stride ld in global, LDST in LDS)
DEV void fill_tile(const bf16* __restrict__ G, int ld, int rowbase, int k0,
                   bf16* lds, int tid) {
#pragma unroll
  for (int t = 0; t < 2; ++t) {
    int c = tid + t * 256;          // 512 chunks of 16B
    int r = c >> 2, o = (c & 3) * 8;
    __builtin_prefetch(G + (size_t)(rowbase + r) * ld + k0 + o + 64, 0, 1);
    copy16_async(G + (size_t)(rowbase + r) * ld + k0 + o, lds + r * LDST + o);
  }
}
DEV v16bf lds_a_frag(const bf16* base, int mrow, int lane) {
  int half = lane >> 4, rl = lane & 15;
  return load_a_pair(base + (mrow + rl) * LDST + half * 16);
}
DEV v16bf lds_b_frag(const bf16* base, int ncol, int lane) {
  int half = lane >> 4, rl = lane & 15;
  return load_bf16x16(base + (ncol + rl) * LDST + half * 16);
}

// ---------------- weight convert + transpose: out[N][K] = bf16(in[K][N]) ----
__global__ void conv_transpose_bf16(const float* __restrict__ in, bf16* __restrict__ out,
                                    int K, int N) {
  size_t idx = (size_t)blockIdx.x * blockDim.x + threadIdx.x;
  if (idx >= (size_t)K * N) return;
  int n = (int)(idx / K), k = (int)(idx % K);
  out[idx] = (bf16)in[(size_t)k * N + n];
}

// ---------------- LayerNorm (optionally fused window-partition gather) ------
template <bool GATHER>
__global__ __launch_bounds__(256) void ln_kernel(const float* __restrict__ x,
                                                 const float* __restrict__ g,
                                                 const float* __restrict__ b,
                                                 bf16* __restrict__ out) {
  int lane = threadIdx.x & 31;
  int row = blockIdx.x * 8 + (threadIdx.x >> 5);
  size_t src;
  if (GATHER) { int t; win_map(row, t); src = (size_t)t * Cdim; }
  else        { src = (size_t)row * Cdim; }
  float vals[12]; float s = 0.f;
#pragma unroll
  for (int i = 0; i < 12; ++i) { vals[i] = x[src + lane + 32 * i]; s += vals[i]; }
#pragma unroll
  for (int off = 16; off; off >>= 1) s += __shfl_xor(s, off, 32);
  float mu = s * (1.0f / Cdim), vs = 0.f;
#pragma unroll
  for (int i = 0; i < 12; ++i) { float d = vals[i] - mu; vs += d * d; }
#pragma unroll
  for (int off = 16; off; off >>= 1) vs += __shfl_xor(vs, off, 32);
  float rinv = rsqrtf(vs * (1.0f / Cdim) + 1e-5f);
#pragma unroll
  for (int i = 0; i < 12; ++i) {
    int c = lane + 32 * i;
    out[(size_t)row * Cdim + c] = (bf16)((vals[i] - mu) * rinv * g[c] + b[c]);
  }
}

// ------- 128x128-tile bf16 WMMA GEMM, double-buffered async LDS staging -----
enum { EPI_QKV = 0, EPI_PROJ = 1, EPI_FC1 = 2, EPI_FC2 = 3 };

template <int EPI>
__global__ __launch_bounds__(256) void gemm_bf16(const bf16* __restrict__ A,
                                                 const bf16* __restrict__ Bt,
                                                 const float* __restrict__ bias,
                                                 int K, int Ncols,
                                                 const float* __restrict__ resid,
                                                 float* __restrict__ outF,
                                                 bf16* __restrict__ outB) {
  __shared__ bf16 sA[2][128 * LDST];
  __shared__ bf16 sB[2][128 * LDST];
  int tid = threadIdx.x, lane = tid & 31, wid = tid >> 5;
  int wm = wid >> 1, wn = wid & 1;        // 4x2 waves, each 32(M) x 64(N)
  int mblock = blockIdx.x * 128, nblock = blockIdx.y * 128;
  int ml = wm * 32, nl = wn * 64;

  v8f acc[2][4] = {};
  int nk = K / 32;
  fill_tile(A,  K, mblock, 0, &sA[0][0], tid);
  fill_tile(Bt, K, nblock, 0, &sB[0][0], tid);
  for (int kt = 0; kt < nk; ++kt) {
    int buf = kt & 1;
    if (kt + 1 < nk) {
      fill_tile(A,  K, mblock, (kt + 1) * 32, &sA[buf ^ 1][0], tid);
      fill_tile(Bt, K, nblock, (kt + 1) * 32, &sB[buf ^ 1][0], tid);
      async_wait<4>();   // async completes in order: current buffer's 4 chunks landed
    } else {
      async_wait<0>();
    }
    __syncthreads();
    v16bf a0 = lds_a_frag(&sA[buf][0], ml,      lane);
    v16bf a1 = lds_a_frag(&sA[buf][0], ml + 16, lane);
    v16bf b0 = lds_b_frag(&sB[buf][0], nl,      lane);
    v16bf b1 = lds_b_frag(&sB[buf][0], nl + 16, lane);
    v16bf b2 = lds_b_frag(&sB[buf][0], nl + 32, lane);
    v16bf b3 = lds_b_frag(&sB[buf][0], nl + 48, lane);
    acc[0][0] = wmma_bf16(a0, b0, acc[0][0]);
    acc[0][1] = wmma_bf16(a0, b1, acc[0][1]);
    acc[0][2] = wmma_bf16(a0, b2, acc[0][2]);
    acc[0][3] = wmma_bf16(a0, b3, acc[0][3]);
    acc[1][0] = wmma_bf16(a1, b0, acc[1][0]);
    acc[1][1] = wmma_bf16(a1, b1, acc[1][1]);
    acc[1][2] = wmma_bf16(a1, b2, acc[1][2]);
    acc[1][3] = wmma_bf16(a1, b3, acc[1][3]);
    __syncthreads();     // all waves done with buf before it is refilled
  }

  int half = lane >> 4, rl = lane & 15;
  float bv[4];
#pragma unroll
  for (int sn = 0; sn < 4; ++sn) bv[sn] = bias[nblock + nl + sn * 16 + rl];
#pragma unroll
  for (int sm = 0; sm < 2; ++sm) {
#pragma unroll
    for (int r = 0; r < 8; ++r) {
      int rowg = mblock + ml + sm * 16 + half * 8 + r;
      size_t rowoff;
      if (EPI == EPI_PROJ) { int t; win_map(rowg, t); rowoff = (size_t)t * Cdim; }
      else                 { rowoff = (size_t)rowg * Ncols; }
#pragma unroll
      for (int sn = 0; sn < 4; ++sn) {
        int colg = nblock + nl + sn * 16 + rl;
        float v = acc[sm][sn][r] + bv[sn];
        if (EPI == EPI_QKV) {
          outB[rowoff + colg] = (bf16)v;
        } else if (EPI == EPI_FC1) {
          float ge = 0.5f * v * (1.0f + erff(v * 0.70710678118654752f));
          outB[rowoff + colg] = (bf16)ge;
        } else { // PROJ / FC2: residual add, fp32 store
          outF[rowoff + colg] = resid[rowoff + colg] + v;
        }
      }
    }
  }
}

// ---------------- windowed attention: 1 wave per (window, head) -------------
static constexpr int PST = 72;  // padded LDS strides (elements)
__global__ __launch_bounds__(128) void attn_kernel(const bf16* __restrict__ qkv,
                                                   bf16* __restrict__ oattn) {
  __shared__ bf16 sP[4][64 * PST];   // softmaxed scores, rows padded to 64
  __shared__ bf16 sVt[4][32 * PST];  // V transposed: [hd][seq(pad 64)]
  int lane = threadIdx.x & 31, w = threadIdx.x >> 5;
  int half = lane >> 4, rl = lane & 15;
  int pair = blockIdx.x * 4 + w;
  int win = pair / NH, hh = pair % NH;
  const bf16* qb = qkv + (size_t)win * Nwin * QKV + hh * 32;
  const bf16* kb = qb + Cdim;
  const bf16* vb = qb + 2 * Cdim;

  // stage V^T into LDS (pad seq>=49 with zero)
  for (int idx = lane; idx < 32 * 64; idx += 32) {
    int d = idx >> 6, seq = idx & 63;
    sVt[w][d * PST + seq] = (seq < Nwin) ? vb[(size_t)seq * QKV + d] : (bf16)0.0f;
  }
  // K as B-fragments (scores = q @ k^T, K-dim = hd = 32, one WMMA step)
  v16bf bk[4];
#pragma unroll
  for (int ni = 0; ni < 4; ++ni) {
    int n = ni * 16 + rl;
    bk[ni] = (n < Nwin) ? load_bf16x16(kb + (size_t)n * QKV + half * 16) : zero_frag();
  }

  const float scale = 0.17677669529663687f; // 1/sqrt(32)
  for (int mi = 0; mi < 4; ++mi) {
    v16bf aq;
    { int row = mi * 16 + rl;
      aq = (row < Nwin) ? load_a_pair(qb + (size_t)row * QKV + half * 16) : zero_frag(); }
    v8f s[4] = {};
#pragma unroll
    for (int ni = 0; ni < 4; ++ni) s[ni] = wmma_bf16(aq, bk[ni], s[ni]);

    // register softmax: C-tile row r lives in lanes {0..15} (M=r) / {16..31} (M=8+r),
    // so a 16-lane xor butterfly (masks 1..8) reduces across the full row.
#pragma unroll
    for (int r = 0; r < 8; ++r) {
      int row = mi * 16 + half * 8 + r;
      float t[4]; float m = -3.0e38f;
#pragma unroll
      for (int ni = 0; ni < 4; ++ni) {
        int col = ni * 16 + rl;
        t[ni] = (col < Nwin) ? s[ni][r] * scale : -3.0e38f;
        m = fmaxf(m, t[ni]);
      }
#pragma unroll
      for (int off = 1; off < 16; off <<= 1) m = fmaxf(m, __shfl_xor(m, off, 32));
      float e[4]; float sum = 0.f;
#pragma unroll
      for (int ni = 0; ni < 4; ++ni) {
        int col = ni * 16 + rl;
        e[ni] = (col < Nwin) ? __expf(t[ni] - m) : 0.0f;
        sum += e[ni];
      }
#pragma unroll
      for (int off = 1; off < 16; off <<= 1) sum += __shfl_xor(sum, off, 32);
      float inv = 1.0f / sum;
#pragma unroll
      for (int ni = 0; ni < 4; ++ni) sP[w][row * PST + ni * 16 + rl] = (bf16)(e[ni] * inv);
    }
  }
  __syncthreads(); // uniform across all 4 waves

  // O = P(64x64) @ V(64x32), K-loop of 2 WMMA steps
  size_t obase = (size_t)win * Nwin * Cdim + hh * 32;
  for (int mo = 0; mo < 4; ++mo) {
    v8f o0 = {}, o1 = {};
#pragma unroll
    for (int k0 = 0; k0 < 64; k0 += 32) {
      v16bf ap  = load_a_pair(&sP[w][(mo * 16 + rl) * PST + k0 + half * 16]);
      v16bf bv0 = load_bf16x16(&sVt[w][(rl)      * PST + k0 + half * 16]);
      v16bf bv1 = load_bf16x16(&sVt[w][(16 + rl) * PST + k0 + half * 16]);
      o0 = wmma_bf16(ap, bv0, o0);
      o1 = wmma_bf16(ap, bv1, o1);
    }
#pragma unroll
    for (int r = 0; r < 8; ++r) {
      int row = mo * 16 + half * 8 + r;
      if (row < Nwin) {
        oattn[obase + (size_t)row * Cdim + rl]      = (bf16)o0[r];
        oattn[obase + (size_t)row * Cdim + 16 + rl] = (bf16)o1[r];
      }
    }
  }
}

// ---------------- host-side orchestration -----------------------------------
extern "C" void kernel_launch(void* const* d_in, const int* in_sizes, int n_in,
                              void* d_out, int out_size, void* d_ws, size_t ws_size,
                              hipStream_t stream) {
  const float* x      = (const float*)d_in[0];
  const float* ln1_g  = (const float*)d_in[1];
  const float* ln1_b  = (const float*)d_in[2];
  const float* w_qkv  = (const float*)d_in[3];
  const float* b_qkv  = (const float*)d_in[4];
  const float* w_proj = (const float*)d_in[5];
  const float* b_proj = (const float*)d_in[6];
  const float* ln2_g  = (const float*)d_in[7];
  const float* ln2_b  = (const float*)d_in[8];
  const float* w_fc1  = (const float*)d_in[9];
  const float* b_fc1  = (const float*)d_in[10];
  const float* w_fc2  = (const float*)d_in[11];
  const float* b_fc2  = (const float*)d_in[12];
  float* out = (float*)d_out;

  char* ws = (char*)d_ws;
  size_t oYW    = 0;
  size_t oQKVB  = oYW    + (size_t)Mrows * Cdim * 2;
  size_t oOATTN = oQKVB  + (size_t)Mrows * QKV  * 2;
  size_t oX2    = oOATTN + (size_t)Mrows * Cdim * 2;
  size_t oG1    = oX2    + (size_t)Mrows * Cdim * 4;
  size_t oWQ    = oG1    + (size_t)Mrows * HID  * 2;
  size_t oWP    = oWQ    + (size_t)QKV  * Cdim * 2;
  size_t oWF1   = oWP    + (size_t)Cdim * Cdim * 2;
  size_t oWF2   = oWF1   + (size_t)HID  * Cdim * 2;

  bf16*  yw     = (bf16*)(ws + oYW);     // also reused as LN2 output
  bf16*  qkvb   = (bf16*)(ws + oQKVB);
  bf16*  oattn  = (bf16*)(ws + oOATTN);
  float* x2     = (float*)(ws + oX2);
  bf16*  g1     = (bf16*)(ws + oG1);
  bf16*  wqkv_t = (bf16*)(ws + oWQ);
  bf16*  wproj_t= (bf16*)(ws + oWP);
  bf16*  wfc1_t = (bf16*)(ws + oWF1);
  bf16*  wfc2_t = (bf16*)(ws + oWF2);

  auto cdiv = [](size_t a, size_t b) { return (unsigned)((a + b - 1) / b); };

  // weights -> bf16, transposed to [N][K]
  conv_transpose_bf16<<<cdiv((size_t)Cdim * QKV, 256), 256, 0, stream>>>(w_qkv,  wqkv_t,  Cdim, QKV);
  conv_transpose_bf16<<<cdiv((size_t)Cdim * Cdim, 256), 256, 0, stream>>>(w_proj, wproj_t, Cdim, Cdim);
  conv_transpose_bf16<<<cdiv((size_t)Cdim * HID, 256), 256, 0, stream>>>(w_fc1,  wfc1_t,  Cdim, HID);
  conv_transpose_bf16<<<cdiv((size_t)HID * Cdim, 256), 256, 0, stream>>>(w_fc2,  wfc2_t,  HID,  Cdim);

  // LN1 + roll + window partition -> bf16
  ln_kernel<true><<<Mrows / 8, 256, 0, stream>>>(x, ln1_g, ln1_b, yw);

  // QKV GEMM: [100352,384] x [384,1152]
  gemm_bf16<EPI_QKV><<<dim3(Mrows / 128, QKV / 128), 256, 0, stream>>>(
      yw, wqkv_t, b_qkv, Cdim, QKV, nullptr, nullptr, qkvb);

  // windowed attention
  attn_kernel<<<(NWINS * NH) / 4, 128, 0, stream>>>(qkvb, oattn);

  // proj GEMM + window reverse/roll scatter + residual -> x2 (fp32)
  gemm_bf16<EPI_PROJ><<<dim3(Mrows / 128, Cdim / 128), 256, 0, stream>>>(
      oattn, wproj_t, b_proj, Cdim, Cdim, x, x2, nullptr);

  // LN2 -> bf16 (reuses yw buffer)
  ln_kernel<false><<<Mrows / 8, 256, 0, stream>>>(x2, ln2_g, ln2_b, yw);

  // FC1 + GELU -> bf16
  gemm_bf16<EPI_FC1><<<dim3(Mrows / 128, HID / 128), 256, 0, stream>>>(
      yw, wfc1_t, b_fc1, Cdim, HID, nullptr, nullptr, g1);

  // FC2 + residual -> out (fp32)
  gemm_bf16<EPI_FC2><<<dim3(Mrows / 128, Cdim / 128), 256, 0, stream>>>(
      g1, wfc2_t, b_fc2, HID, Cdim, x2, out, nullptr);

  (void)in_sizes; (void)n_in; (void)out_size; (void)ws_size;
}
